// GFGCN_4982162063661
// MI455X (gfx1250) — compile-verified, matrix operand order checked by reading, not run on hardware
//
#include <hip/hip_runtime.h>

typedef __attribute__((ext_vector_type(16))) __bf16 v16bf;
typedef __attribute__((ext_vector_type(8)))  __bf16 v8bf;
typedef __attribute__((ext_vector_type(8)))  float  v8f;

#define BM 128
#define BN 64
#define BK 32

// CDNA5 async memory->LDS staging (ASYNCcnt-tracked, bypasses VGPRs)
#define ASYNC_LOAD_B128(ldsoff, gaddr)                                   \
  asm volatile("global_load_async_to_lds_b128 %0, %1, off"               \
               :: "v"(ldsoff), "v"(gaddr) : "memory")
#define WAIT_ASYNC() asm volatile("s_wait_asynccnt 0x0" ::: "memory")

// ---------------------------------------------------------------------------
// f32 -> bf16 conversion, row-major (8 elements/thread)
// ---------------------------------------------------------------------------
__global__ __launch_bounds__(256) void cvt_f32_bf16(const float* __restrict__ in,
                                                    __bf16* __restrict__ out,
                                                    long n) {
  long i = (long)blockIdx.x * blockDim.x + threadIdx.x;
  long base = i * 8;
  if (base + 8 <= n) {
    float4 a = ((const float4*)in)[i * 2 + 0];
    float4 b = ((const float4*)in)[i * 2 + 1];
    v8bf o;
    o[0] = (__bf16)a.x; o[1] = (__bf16)a.y; o[2] = (__bf16)a.z; o[3] = (__bf16)a.w;
    o[4] = (__bf16)b.x; o[5] = (__bf16)b.y; o[6] = (__bf16)b.z; o[7] = (__bf16)b.w;
    *(v8bf*)(out + base) = o;
  } else {
    for (long j = base; j < n; ++j) out[j] = (__bf16)in[j];
  }
}

// f32 [R][C] row-major  ->  bf16 column-major ([C][R])
__global__ __launch_bounds__(256) void cvt_f32_bf16_t(const float* __restrict__ in,
                                                      __bf16* __restrict__ out,
                                                      int R, int C) {
  int idx = blockIdx.x * blockDim.x + threadIdx.x;
  if (idx < R * C) {
    int r = idx / C, c = idx % C;
    out[(long)c * R + r] = (__bf16)in[idx];
  }
}

// ---------------------------------------------------------------------------
// bf16 WMMA GEMM:  C[M,N] = A[M,K] @ B[K,N]
//   A row-major bf16 (lda=K); B **column-major** bf16 (element (k,n) at n*K+k)
// mode 0: store (bf16)C column-major (consumed as next B / as T1,T2)
// mode 1: store (bf16)relu(h2*C + h0*T1 + h1*T2 + bias) row-major
// mode 2: store f32   (h2*C + h0*T1 + h1*T2 + bias)     row-major
// T1,T2 are column-major bf16.
// Block tile 128x64x32, 8 waves (4Mx2N), wave = 2x2 WMMA 16x16 tiles.
// Double-buffered LDS, tiles staged with global_load_async_to_lds_b128.
// ---------------------------------------------------------------------------
union Frag { v16bf v; v8bf h[2]; };

__global__ __launch_bounds__(256) void wmma_gemm_bf16(
    const __bf16* __restrict__ A, const __bf16* __restrict__ B,
    int M, int N, int K,
    const __bf16* __restrict__ T1, const __bf16* __restrict__ T2,
    const float* __restrict__ hvec, const float* __restrict__ bias,
    int mode, __bf16* __restrict__ outb, float* __restrict__ outf) {

  __shared__ __bf16 As[2][BM * BK];   // 2 x 8 KB, row-major [BM][BK]
  __shared__ __bf16 Bs[2][BN * BK];   // 2 x 4 KB, [n][k] (B is column-major)

  const int tid  = threadIdx.x;
  const int lane = tid & 31;
  const int wave = tid >> 5;
  const int wm   = wave & 3;
  const int wn   = wave >> 2;
  const int m0   = blockIdx.x * BM;
  const int n0   = blockIdx.y * BN;

  const int half = lane >> 4;
  const int lidx = lane & 15;

  // per-thread staging coordinates (all tiles are exact multiples, no guards)
  const int arow = tid >> 2;            // used twice with +64: A rows
  const int acol = (tid & 3) * 8;       // 16B chunk within a 64B row
  const int brow = tid >> 2;            // B (column-major) row: 0..63
  const int bcol = (tid & 3) * 8;

  auto issueA = [&](int buf, int kt) {
#pragma unroll
    for (int j = 0; j < 2; ++j) {
      int r = arow + 64 * j;
      unsigned lo = (unsigned)(uintptr_t)(void*)&As[buf][r * BK + acol];
      const void* gp = (const void*)&A[(long)(m0 + r) * K + kt + acol];
      ASYNC_LOAD_B128(lo, gp);
    }
  };
  auto issueB = [&](int buf, int kt) {
    unsigned lo = (unsigned)(uintptr_t)(void*)&Bs[buf][brow * BK + bcol];
    const void* gp = (const void*)&B[(long)(n0 + brow) * K + kt + bcol];
    ASYNC_LOAD_B128(lo, gp);
  };

  v8f acc[2][2];
#pragma unroll
  for (int mt = 0; mt < 2; ++mt)
#pragma unroll
    for (int nt = 0; nt < 2; ++nt) acc[mt][nt] = (v8f)0.0f;

  // prologue: stage tile 0
  issueA(0, 0);
  issueB(0, 0);
  WAIT_ASYNC();
  __syncthreads();

  const int nk = K / BK;
  for (int it = 0; it < nk; ++it) {
    const int cur = it & 1, nxt = cur ^ 1;
    const bool more = (it + 1) < nk;
    if (more) {                       // overlap next tile fetch with compute
      issueA(nxt, (it + 1) * BK);
      issueB(nxt, (it + 1) * BK);
    }

    // Fragments per CDNA5 16-bit 16x32 layout:
    // lanes 0-15: K 0-7 / 16-23, lanes 16-31: K 8-15 / 24-31 (packed pairs)
    Frag af[2], bf[2];
#pragma unroll
    for (int mt = 0; mt < 2; ++mt) {
      int row = wm * 32 + mt * 16 + lidx;
      af[mt].h[0] = *(v8bf*)(&As[cur][row * BK + 8 * half]);
      af[mt].h[1] = *(v8bf*)(&As[cur][row * BK + 16 + 8 * half]);
    }
#pragma unroll
    for (int nt = 0; nt < 2; ++nt) {
      int col = wn * 32 + nt * 16 + lidx;
      bf[nt].h[0] = *(v8bf*)(&Bs[cur][col * BK + 8 * half]);
      bf[nt].h[1] = *(v8bf*)(&Bs[cur][col * BK + 16 + 8 * half]);
    }

#pragma unroll
    for (int mt = 0; mt < 2; ++mt)
#pragma unroll
      for (int nt = 0; nt < 2; ++nt)
        acc[mt][nt] = __builtin_amdgcn_wmma_f32_16x16x32_bf16(
            false, af[mt].v, false, bf[nt].v,
            (short)0, acc[mt][nt], false, false);

    if (more) WAIT_ASYNC();           // next tile resident before the barrier
    __syncthreads();
  }

  // epilogue: C/D layout -> lane lidx = N, VGPR i = M row (i + 8*half)
  float h0 = 0.f, h1 = 0.f, h2 = 1.f;
  if (mode != 0) { h0 = hvec[0]; h1 = hvec[1]; h2 = hvec[2]; }

#pragma unroll
  for (int mt = 0; mt < 2; ++mt) {
#pragma unroll
    for (int nt = 0; nt < 2; ++nt) {
      int col = n0 + wn * 32 + nt * 16 + lidx;
#pragma unroll
      for (int i = 0; i < 8; ++i) {
        int row = m0 + wm * 32 + mt * 16 + half * 8 + i;
        float v = acc[mt][nt][i];
        if (mode == 0) {
          outb[(long)col * M + row] = (__bf16)v;          // column-major
        } else {
          long tOff = (long)col * M + row;                // T1/T2 column-major
          float r = h2 * v + h0 * (float)T1[tOff] + h1 * (float)T2[tOff]
                    + bias[col];
          if (mode == 1) {
            r = fmaxf(r, 0.0f);
            outb[(long)row * N + col] = (__bf16)r;        // row-major
          } else {
            outf[(long)row * N + col] = r;
          }
        }
      }
    }
  }
}

// ---------------------------------------------------------------------------
// Host-side orchestration
// ---------------------------------------------------------------------------
static inline void launch_cvt(const float* in, __bf16* out, long n, hipStream_t s) {
  long threads = (n + 7) / 8;
  int blocks = (int)((threads + 255) / 256);
  hipLaunchKernelGGL(cvt_f32_bf16, dim3(blocks), dim3(256), 0, s, in, out, n);
}

static inline void launch_cvt_t(const float* in, __bf16* out, int R, int C,
                                hipStream_t s) {
  int blocks = (R * C + 255) / 256;
  hipLaunchKernelGGL(cvt_f32_bf16_t, dim3(blocks), dim3(256), 0, s, in, out, R, C);
}

static inline void launch_gemm(const __bf16* A, const __bf16* B, int M, int N, int K,
                               const __bf16* T1, const __bf16* T2,
                               const float* hv, const float* bias, int mode,
                               __bf16* ob, float* of, hipStream_t s) {
  dim3 grid(M / BM, N / BN), blk(256);
  hipLaunchKernelGGL(wmma_gemm_bf16, grid, blk, 0, s,
                     A, B, M, N, K, T1, T2, hv, bias, mode, ob, of);
}

extern "C" void kernel_launch(void* const* d_in, const int* in_sizes, int n_in,
                              void* d_out, int out_size, void* d_ws, size_t ws_size,
                              hipStream_t stream) {
  (void)in_sizes; (void)n_in; (void)out_size; (void)ws_size;
  const float* S  = (const float*)d_in[0];
  const float* X  = (const float*)d_in[1];
  const float* W1 = (const float*)d_in[2];
  const float* h1 = (const float*)d_in[3];
  const float* b1 = (const float*)d_in[4];
  const float* W2 = (const float*)d_in[5];
  const float* h2 = (const float*)d_in[6];
  const float* b2 = (const float*)d_in[7];
  const float* W3 = (const float*)d_in[8];
  const float* h3 = (const float*)d_in[9];
  const float* b3 = (const float*)d_in[10];
  float* out = (float*)d_out;

  const int N = 8192, IND = 512, HID = 128, OUTD = 64;

  char* ws = (char*)d_ws;
  size_t off = 0;
  auto alloc = [&](size_t bytes) -> void* {
    void* p = ws + off;
    off += (bytes + 255) & ~(size_t)255;
    return p;
  };
  __bf16* Sb  = (__bf16*)alloc((size_t)N * N * 2);      // row-major, 134 MB
  __bf16* Xb  = (__bf16*)alloc((size_t)N * IND * 2);    // row-major
  __bf16* W1t = (__bf16*)alloc((size_t)IND * HID * 2);  // column-major
  __bf16* W2t = (__bf16*)alloc((size_t)HID * HID * 2);  // column-major
  __bf16* W3t = (__bf16*)alloc((size_t)HID * OUTD * 2); // column-major
  __bf16* Y0  = (__bf16*)alloc((size_t)N * HID * 2);    // column-major
  __bf16* Z1  = (__bf16*)alloc((size_t)N * HID * 2);    // column-major
  __bf16* H1  = (__bf16*)alloc((size_t)N * HID * 2);    // row-major
  __bf16* H2  = (__bf16*)alloc((size_t)N * HID * 2);    // row-major

  launch_cvt(S, Sb, (long)N * N, stream);
  launch_cvt(X, Xb, (long)N * IND, stream);
  launch_cvt_t(W1, W1t, IND, HID, stream);
  launch_cvt_t(W2, W2t, HID, HID, stream);
  launch_cvt_t(W3, W3t, HID, OUTD, stream);

  // ---- layer 1: relu(h0*Y0 + h1*S Y0 + h2*S(S Y0) + b1) ----
  launch_gemm(Xb, W1t, N, HID, IND, nullptr, nullptr, nullptr, nullptr, 0, Y0, nullptr, stream);
  launch_gemm(Sb, Y0,  N, HID, N,   nullptr, nullptr, nullptr, nullptr, 0, Z1, nullptr, stream);
  launch_gemm(Sb, Z1,  N, HID, N,   Y0, Z1, h1, b1, 1, H1, nullptr, stream);

  // ---- layer 2 ----
  launch_gemm(H1, W2t, N, HID, HID, nullptr, nullptr, nullptr, nullptr, 0, Y0, nullptr, stream);
  launch_gemm(Sb, Y0,  N, HID, N,   nullptr, nullptr, nullptr, nullptr, 0, Z1, nullptr, stream);
  launch_gemm(Sb, Z1,  N, HID, N,   Y0, Z1, h2, b2, 1, H2, nullptr, stream);

  // ---- layer 3 (identity activation, f32 output) ----
  launch_gemm(H2, W3t, N, OUTD, HID, nullptr, nullptr, nullptr, nullptr, 0, Y0, nullptr, stream);
  launch_gemm(Sb, Y0,  N, OUTD, N,   nullptr, nullptr, nullptr, nullptr, 0, Z1, nullptr, stream);
  launch_gemm(Sb, Z1,  N, OUTD, N,   Y0, Z1, h3, b3, 2, nullptr, out, stream);
}
